// NeuronMoonViTEncoderLayer_38019050504674
// MI455X (gfx1250) — compile-verified, matrix operand order checked by reading, not run on hardware
//
#include <hip/hip_runtime.h>
#include <hip/hip_bf16.h>
#include <math.h>

// ---------------- problem constants ----------------
#define SEQ     3072
#define HIDDEN  1152
#define MLP     4304
#define NHEADS  16
#define HDIM    72
#define HPAD    96          // head dim padded to multiple of 32 for WMMA K
#define QKV_N   (3*HIDDEN)  // 3456
#define LN_EPS  1e-5f

typedef __bf16 bf16;
typedef __attribute__((ext_vector_type(16))) __bf16 v16bf;
typedef __attribute__((ext_vector_type(8)))  float  v8f;
typedef __attribute__((ext_vector_type(4)))  int    v4i;

// ---------------- CDNA5 async global->LDS copy (guarded) ----------------
#if defined(__has_builtin)
#if __has_builtin(__builtin_amdgcn_global_load_async_to_lds_b128)
#define HAVE_ASYNC_LDS 1
#endif
#endif
#ifndef HAVE_ASYNC_LDS
#define HAVE_ASYNC_LDS 0
#endif

#if HAVE_ASYNC_LDS
typedef __attribute__((address_space(1))) v4i* gptr_v4i;
typedef __attribute__((address_space(3))) v4i* lptr_v4i;
#endif

// copy 16 bytes global -> LDS (per-lane). Falls back to sync load+ds_store.
static __device__ __forceinline__ void stage16(const bf16* g, bf16* l) {
#if HAVE_ASYNC_LDS
  // flat LDS pointer: low 32 bits are the LDS byte offset (aperture truncation);
  // flat global pointer: value == AS1 address.
  gptr_v4i gp = (gptr_v4i)(uintptr_t)g;
  lptr_v4i lp = (lptr_v4i)(unsigned int)(uintptr_t)l;
  __builtin_amdgcn_global_load_async_to_lds_b128(gp, lp, 0, 0);
#else
  *(uint4*)l = *(const uint4*)g;
#endif
}

static __device__ __forceinline__ void stage_zero16(bf16* l) {
  *(uint4*)l = make_uint4(0u, 0u, 0u, 0u);
}

static __device__ __forceinline__ void stage_wait() {
#if HAVE_ASYNC_LDS
#if defined(__has_builtin) && __has_builtin(__builtin_amdgcn_s_wait_asynccnt)
  __builtin_amdgcn_s_wait_asynccnt(0);
#else
  asm volatile("s_wait_asynccnt 0" ::: "memory");
#endif
#endif
}

static __device__ __forceinline__ v8f wmma_bf16(v16bf a, v16bf b, v8f c) {
  // (neg_a, A, neg_b, B, c_mod, C, reuse_a, reuse_b)
  return __builtin_amdgcn_wmma_f32_16x16x32_bf16(false, a, false, b, (short)0, c,
                                                 false, false);
}

union FragU { uint4 u[2]; v16bf v; };

// Load a 16x32 bf16 operand fragment (A-layout; identical pattern used for the
// NT B operand since its rows are also K-contiguous).
// lane<16 : row=lane,    K = {0..7, 16..23}
// lane>=16: row=lane-16, K = {8..15, 24..31}
static __device__ __forceinline__ v16bf load_frag(const bf16* base, int ld) {
  const int lane = threadIdx.x & 31;
  const int r    = lane & 15;
  const int k8   = (lane >> 4) * 8;
  FragU f;
  f.u[0] = *(const uint4*)(base + r * ld + k8);
  f.u[1] = *(const uint4*)(base + r * ld + k8 + 16);
  return f.v;
}

// ---------------- f32 -> bf16 convert ----------------
__global__ void f32_to_bf16_kernel(const float* __restrict__ in,
                                   bf16* __restrict__ out, int n) {
  int i = blockIdx.x * blockDim.x + threadIdx.x;
  if (i < n) out[i] = (bf16)in[i];
}

// ---------------- LayerNorm (one block per row) ----------------
__global__ __launch_bounds__(256)
void ln_kernel(const float* __restrict__ x, const float* __restrict__ g,
               const float* __restrict__ b, bf16* __restrict__ out, int N) {
  __shared__ float red[256];
  const int row = blockIdx.x;
  const int tid = threadIdx.x;
  const float* xr = x + (size_t)row * N;

  float s = 0.f, s2 = 0.f;
  for (int i = tid; i < N; i += 256) { float v = xr[i]; s += v; s2 += v * v; }

  red[tid] = s; __syncthreads();
  for (int o = 128; o > 0; o >>= 1) { if (tid < o) red[tid] += red[tid + o]; __syncthreads(); }
  const float mean = red[0] / (float)N;
  __syncthreads();
  red[tid] = s2; __syncthreads();
  for (int o = 128; o > 0; o >>= 1) { if (tid < o) red[tid] += red[tid + o]; __syncthreads(); }
  const float var  = red[0] / (float)N - mean * mean;
  const float rstd = rsqrtf(var + LN_EPS);

  for (int i = tid; i < N; i += 256)
    out[(size_t)row * N + i] = (bf16)((xr[i] - mean) * rstd * g[i] + b[i]);
}

// ---------------- RoPE + pack q/k/v into padded bf16 [head][SEQ][HPAD] ----------------
__global__ __launch_bounds__(256)
void rope_pack_kernel(const float* __restrict__ qkv, const float* __restrict__ cosb,
                      const float* __restrict__ sinb, bf16* __restrict__ qb,
                      bf16* __restrict__ kb, bf16* __restrict__ vb) {
  const int s = blockIdx.x;
  const float scale = 0.11785113019775792f;  // 1/sqrt(72), folded into q
  for (int idx = threadIdx.x; idx < NHEADS * (HPAD / 2); idx += 256) {
    const int h = idx / (HPAD / 2);
    const int p = idx % (HPAD / 2);
    const size_t o = (size_t)h * SEQ * HPAD + (size_t)s * HPAD + 2 * p;
    if (p < HDIM / 2) {
      const float c  = cosb[s * (HDIM / 2) + p];
      const float sn = sinb[s * (HDIM / 2) + p];
      const float* base = qkv + (size_t)s * QKV_N + h * HDIM + 2 * p;
      const float qr = base[0],          qi = base[1];
      const float kr = base[HIDDEN],     ki = base[HIDDEN + 1];
      const float vr = base[2 * HIDDEN], vi = base[2 * HIDDEN + 1];
      qb[o]     = (bf16)((qr * c - qi * sn) * scale);
      qb[o + 1] = (bf16)((qr * sn + qi * c) * scale);
      kb[o]     = (bf16)(kr * c - ki * sn);
      kb[o + 1] = (bf16)(kr * sn + ki * c);
      vb[o]     = (bf16)vr;
      vb[o + 1] = (bf16)vi;
    } else {  // zero pad dims [72,96)
      qb[o] = (bf16)0.f; qb[o + 1] = (bf16)0.f;
      kb[o] = (bf16)0.f; kb[o + 1] = (bf16)0.f;
      vb[o] = (bf16)0.f; vb[o + 1] = (bf16)0.f;
    }
  }
}

// ---------------- generic NT WMMA GEMM: out[m,n] = sum_k A[m,k]*B[n,k] ----------------
// 256 threads = 8 waves; block tile 128(M) x 64(N); wave tile 32x32 (2x2 WMMA).
// K staged in slabs of 64 via async global->LDS; 8 WMMAs per stage per wave.
// EPI: 0 = +bias -> f32 ; 1 = +bias +resid -> f32 ; 2 = gelu(+bias) -> bf16
#define KSLAB 64
#define LDPG  72  // padded LDS K-stride (bank-conflict free, keeps 16B alignment)
template <int EPI>
__global__ __launch_bounds__(256)
void gemm_nt_kernel(const bf16* __restrict__ A, const bf16* __restrict__ B,
                    const float* __restrict__ bias, const float* __restrict__ resid,
                    void* __restrict__ outv, int M, int N, int K) {
  __shared__ bf16 As[128 * LDPG];  // 18 KB
  __shared__ bf16 Bs[64 * LDPG];   //  9 KB

  const int tid  = threadIdx.x;
  const int wave = tid >> 5;
  const int lane = tid & 31;
  const int m0   = blockIdx.y * 128;
  const int n0   = blockIdx.x * 64;
  const int wm   = wave & 3;   // 4 waves along M
  const int wn   = wave >> 2;  // 2 waves along N

  const v8f vzero = {0.f, 0.f, 0.f, 0.f, 0.f, 0.f, 0.f, 0.f};
  v8f acc[2][2] = {{vzero, vzero}, {vzero, vzero}};

  for (int k0 = 0; k0 < K; k0 += KSLAB) {
    __syncthreads();
    // stage A: 128 rows x 64 K = 1024 16B chunks, 4 per thread
#pragma unroll
    for (int j = 0; j < 4; ++j) {
      const int c  = tid + 256 * j;
      const int r  = c >> 3;
      const int k8 = (c & 7) * 8;
      bf16* l = &As[r * LDPG + k8];
      if (k0 + k8 < K) stage16(A + (size_t)(m0 + r) * K + k0 + k8, l);
      else             stage_zero16(l);
    }
    // stage B: 64 rows x 64 K = 512 chunks, 2 per thread
#pragma unroll
    for (int j = 0; j < 2; ++j) {
      const int c  = tid + 256 * j;
      const int r  = c >> 3;
      const int k8 = (c & 7) * 8;
      bf16* l = &Bs[r * LDPG + k8];
      if ((n0 + r) < N && (k0 + k8) < K) stage16(B + (size_t)(n0 + r) * K + k0 + k8, l);
      else                               stage_zero16(l);
    }
    stage_wait();          // own async transfers landed in LDS
    __syncthreads();       // everyone's transfers visible

#pragma unroll
    for (int ks = 0; ks < 2; ++ks) {
      const v16bf a0 = load_frag(As + (wm * 32) * LDPG + ks * 32, LDPG);
      const v16bf a1 = load_frag(As + (wm * 32 + 16) * LDPG + ks * 32, LDPG);
      const v16bf b0 = load_frag(Bs + (wn * 32) * LDPG + ks * 32, LDPG);
      const v16bf b1 = load_frag(Bs + (wn * 32 + 16) * LDPG + ks * 32, LDPG);
      acc[0][0] = wmma_bf16(a0, b0, acc[0][0]);
      acc[0][1] = wmma_bf16(a0, b1, acc[0][1]);
      acc[1][0] = wmma_bf16(a1, b0, acc[1][0]);
      acc[1][1] = wmma_bf16(a1, b1, acc[1][1]);
    }
  }

  // epilogue. C layout: vgpr i -> row (lane<16 ? i : i+8), col = lane&15
  const int colb = lane & 15;
  const int rowb = (lane >> 4) * 8;
#pragma unroll
  for (int ti = 0; ti < 2; ++ti)
#pragma unroll
    for (int tj = 0; tj < 2; ++tj) {
      const int n = n0 + wn * 32 + tj * 16 + colb;
      if (n >= N) continue;
      const float bv = bias[n];
#pragma unroll
      for (int i = 0; i < 8; ++i) {
        const int m = m0 + wm * 32 + ti * 16 + rowb + i;
        float x = acc[ti][tj][i] + bv;
        if (EPI == 1) x += resid[(size_t)m * N + n];
        if (EPI == 2) {
          const float t = tanhf(0.7978845608028654f * (x + 0.044715f * x * x * x));
          ((bf16*)outv)[(size_t)m * N + n] = (bf16)(0.5f * x * (1.f + t));
        } else {
          ((float*)outv)[(size_t)m * N + n] = x;
        }
      }
    }
}

// ---------------- flash attention (bf16 WMMA, online softmax) ----------------
// grid: (SEQ/128, NHEADS); 8 waves/block; each wave owns 16 query rows.
#define LDP 40
__global__ __launch_bounds__(256)
void attn_kernel(const bf16* __restrict__ qb, const bf16* __restrict__ kb,
                 const bf16* __restrict__ vb, bf16* __restrict__ out) {
  __shared__ bf16 Vt[HPAD * LDP];       // V^T per key block: [dim][key]
  __shared__ bf16 Pb[8 * 16 * LDP];     // per-wave probability tile 16x32

  const int head = blockIdx.y;
  const int tid  = threadIdx.x;
  const int wave = tid >> 5;
  const int lane = tid & 31;
  const int colb = lane & 15;
  const int rowb = (lane >> 4) * 8;
  const int q0   = blockIdx.x * 128 + wave * 16;

  const bf16* qh = qb + (size_t)head * SEQ * HPAD;
  const bf16* kh = kb + (size_t)head * SEQ * HPAD;
  const bf16* vh = vb + (size_t)head * SEQ * HPAD;

  // Q fragments for this wave's 16 rows (K = 96 -> 3 frags), loaded once.
  v16bf qf[3];
#pragma unroll
  for (int f = 0; f < 3; ++f) qf[f] = load_frag(qh + (size_t)q0 * HPAD + f * 32, HPAD);

  const v8f vzero = {0.f, 0.f, 0.f, 0.f, 0.f, 0.f, 0.f, 0.f};
  v8f O[6];
#pragma unroll
  for (int f = 0; f < 6; ++f) O[f] = vzero;
  float mrow[8], lrow[8];
#pragma unroll
  for (int i = 0; i < 8; ++i) { mrow[i] = -1e30f; lrow[i] = 0.f; }

  bf16* Pw = Pb + wave * 16 * LDP;

  for (int kb0 = 0; kb0 < SEQ; kb0 += 32) {
    __syncthreads();  // protect Vt reuse
    // stage V^T (32 keys x 96 dims -> [dim][key])
    for (int i = tid; i < 32 * HPAD; i += 256) {
      const int key = i / HPAD, d = i % HPAD;
      Vt[d * LDP + key] = vh[(size_t)(kb0 + key) * HPAD + d];
    }
    __syncthreads();

    // scores: 16 rows x 32 keys (two 16x16 tiles, K over padded head dim)
    v8f s0 = vzero, s1 = vzero;
#pragma unroll
    for (int f = 0; f < 3; ++f) {
      const v16bf bk0 = load_frag(kh + (size_t)kb0 * HPAD + f * 32, HPAD);
      const v16bf bk1 = load_frag(kh + (size_t)(kb0 + 16) * HPAD + f * 32, HPAD);
      s0 = wmma_bf16(qf[f], bk0, s0);
      s1 = wmma_bf16(qf[f], bk1, s1);
    }

    // online softmax (row stats live per-lane for the 8 rows of this half)
    float alpha[8];
#pragma unroll
    for (int i = 0; i < 8; ++i) {
      float v = fmaxf(s0[i], s1[i]);
      v = fmaxf(v, __shfl_xor(v, 1, 32));
      v = fmaxf(v, __shfl_xor(v, 2, 32));
      v = fmaxf(v, __shfl_xor(v, 4, 32));
      v = fmaxf(v, __shfl_xor(v, 8, 32));
      const float mn = fmaxf(mrow[i], v);
      alpha[i] = __expf(mrow[i] - mn);
      mrow[i] = mn;
    }
#pragma unroll
    for (int i = 0; i < 8; ++i) {
      s0[i] = __expf(s0[i] - mrow[i]);
      s1[i] = __expf(s1[i] - mrow[i]);
      float v = s0[i] + s1[i];
      v += __shfl_xor(v, 1, 32);
      v += __shfl_xor(v, 2, 32);
      v += __shfl_xor(v, 4, 32);
      v += __shfl_xor(v, 8, 32);
      lrow[i] = lrow[i] * alpha[i] + v;
    }
#pragma unroll
    for (int f = 0; f < 6; ++f)
#pragma unroll
      for (int i = 0; i < 8; ++i) O[f][i] *= alpha[i];

    // P: C-layout f32 -> A-layout bf16 via wave-private LDS
#pragma unroll
    for (int i = 0; i < 8; ++i) {
      Pw[(rowb + i) * LDP + colb]      = (bf16)s0[i];
      Pw[(rowb + i) * LDP + 16 + colb] = (bf16)s1[i];
    }
    asm volatile("s_wait_dscnt 0" ::: "memory");  // wave-local store->load order
    const v16bf pa = load_frag(Pw, LDP);
#pragma unroll
    for (int f = 0; f < 6; ++f) {
      const v16bf bv = load_frag(Vt + (f * 16) * LDP, LDP);
      O[f] = wmma_bf16(pa, bv, O[f]);
    }
  }

  // normalize and emit the real 72 dims into [SEQ][HIDDEN]
  float inv[8];
#pragma unroll
  for (int i = 0; i < 8; ++i) inv[i] = 1.f / lrow[i];
#pragma unroll
  for (int f = 0; f < 6; ++f) {
    const int d = f * 16 + colb;
    if (d < HDIM) {
#pragma unroll
      for (int i = 0; i < 8; ++i)
        out[(size_t)(q0 + rowb + i) * HIDDEN + head * HDIM + d] = (bf16)(O[f][i] * inv[i]);
    }
  }
}

// ---------------- host launch ----------------
extern "C" void kernel_launch(void* const* d_in, const int* in_sizes, int n_in,
                              void* d_out, int out_size, void* d_ws, size_t ws_size,
                              hipStream_t stream) {
  (void)in_sizes; (void)n_in; (void)out_size; (void)ws_size;

  const float* hidden  = (const float*)d_in[0];
  const float* cosb    = (const float*)d_in[1];
  const float* sinb    = (const float*)d_in[2];
  const float* ln0_g   = (const float*)d_in[3];
  const float* ln0_b   = (const float*)d_in[4];
  const float* ln1_g   = (const float*)d_in[5];
  const float* ln1_b   = (const float*)d_in[6];
  const float* wqkv_w  = (const float*)d_in[7];
  const float* wqkv_b  = (const float*)d_in[8];
  const float* wo_w    = (const float*)d_in[9];
  const float* wo_b    = (const float*)d_in[10];
  const float* fc0_w   = (const float*)d_in[11];
  const float* fc0_b   = (const float*)d_in[12];
  const float* fc1_w   = (const float*)d_in[13];
  const float* fc1_b   = (const float*)d_in[14];

  // workspace carve-up (256B aligned)
  uint8_t* w = (uint8_t*)d_ws;
  size_t off = 0;
  auto alloc = [&](size_t bytes) -> void* {
    void* p = w + off;
    off = (off + bytes + 255) & ~(size_t)255;
    return p;
  };
  bf16*  lnbuf   = (bf16*)alloc((size_t)SEQ * HIDDEN * 2);        // ln0, reused for ln1
  bf16*  wqkv_bf = (bf16*)alloc((size_t)QKV_N * HIDDEN * 2);
  bf16*  wo_bf   = (bf16*)alloc((size_t)HIDDEN * HIDDEN * 2);
  bf16*  fc0_bf  = (bf16*)alloc((size_t)MLP * HIDDEN * 2);
  bf16*  fc1_bf  = (bf16*)alloc((size_t)HIDDEN * MLP * 2);
  void*  big     = alloc((size_t)SEQ * QKV_N * 4);                // qkv f32, reused as fc0 out
  float* qkvf    = (float*)big;
  bf16*  fc0out  = (bf16*)big;
  bf16*  qbuf    = (bf16*)alloc((size_t)NHEADS * SEQ * HPAD * 2);
  bf16*  kbuf    = (bf16*)alloc((size_t)NHEADS * SEQ * HPAD * 2);
  bf16*  vbuf    = (bf16*)alloc((size_t)NHEADS * SEQ * HPAD * 2);
  bf16*  attnout = (bf16*)alloc((size_t)SEQ * HIDDEN * 2);
  float* h1      = (float*)alloc((size_t)SEQ * HIDDEN * 4);

  // 1) weights -> bf16
  {
    const int n1 = QKV_N * HIDDEN, n2 = HIDDEN * HIDDEN, n3 = MLP * HIDDEN;
    f32_to_bf16_kernel<<<(n1 + 255) / 256, 256, 0, stream>>>(wqkv_w, wqkv_bf, n1);
    f32_to_bf16_kernel<<<(n2 + 255) / 256, 256, 0, stream>>>(wo_w,   wo_bf,   n2);
    f32_to_bf16_kernel<<<(n3 + 255) / 256, 256, 0, stream>>>(fc0_w,  fc0_bf,  n3);
    f32_to_bf16_kernel<<<(n3 + 255) / 256, 256, 0, stream>>>(fc1_w,  fc1_bf,  n3);
  }
  // 2) LN0
  ln_kernel<<<SEQ, 256, 0, stream>>>(hidden, ln0_g, ln0_b, lnbuf, HIDDEN);
  // 3) QKV = ln0 @ Wqkv^T + b  (f32 out)
  gemm_nt_kernel<0><<<dim3(QKV_N / 64, SEQ / 128), 256, 0, stream>>>(
      lnbuf, wqkv_bf, wqkv_b, nullptr, qkvf, SEQ, QKV_N, HIDDEN);
  // 4) RoPE + pack (scale folded into q)
  rope_pack_kernel<<<SEQ, 256, 0, stream>>>(qkvf, cosb, sinb, qbuf, kbuf, vbuf);
  // 5) flash attention -> attnout bf16 [SEQ][HIDDEN]
  attn_kernel<<<dim3(SEQ / 128, NHEADS), 256, 0, stream>>>(qbuf, kbuf, vbuf, attnout);
  // 6) h1 = hidden + attnout @ Wo^T + b
  gemm_nt_kernel<1><<<dim3(HIDDEN / 64, SEQ / 128), 256, 0, stream>>>(
      attnout, wo_bf, wo_b, hidden, h1, SEQ, HIDDEN, HIDDEN);
  // 7) LN1
  ln_kernel<<<SEQ, 256, 0, stream>>>(h1, ln1_g, ln1_b, lnbuf, HIDDEN);
  // 8) fc0 + gelu -> bf16
  gemm_nt_kernel<2><<<dim3((MLP + 63) / 64, SEQ / 128), 256, 0, stream>>>(
      lnbuf, fc0_bf, fc0_b, nullptr, fc0out, SEQ, MLP, HIDDEN);
  // 9) out = h1 + fc0out @ fc1^T + b  (f32)
  gemm_nt_kernel<1><<<dim3(HIDDEN / 64, SEQ / 128), 256, 0, stream>>>(
      fc0out, fc1_bf, fc1_b, h1, (float*)d_out, SEQ, HIDDEN, MLP);
}
